// BottleneckTransformer_46566035423958
// MI455X (gfx1250) — compile-verified, matrix operand order checked by reading, not run on hardware
//
#include <hip/hip_runtime.h>
#include <hip/hip_bf16.h>

// ---------------- problem constants ----------------
#define BB 16
#define CC 256
#define NS 1024            // H*W = 32*32
#define HEADS 4
#define DH 64              // CC/HEADS

typedef __attribute__((ext_vector_type(16))) _Float16 v16h;
typedef __attribute__((ext_vector_type(8)))  float    v8f;

union Frag16 {
    v16h v;
    uint4 q[2];
    _Float16 h[16];
};

__device__ __forceinline__ v8f wmma_f16(const Frag16& a, const Frag16& b, v8f c) {
    return __builtin_amdgcn_wmma_f32_16x16x32_f16(false, a.v, false, b.v,
                                                  (short)0, c, false, false);
}

// ---------------- workspace layout (bytes) ----------------
constexpr size_t ELE      = (size_t)BB * CC * NS;       // 4,194,304 elements
constexpr size_t OFF_X16  = 0;                          // x as f16  [B,C,N]
constexpr size_t OFF_Y16  = OFF_X16 + ELE * 2;          // conv1+bn+relu out (f16)
constexpr size_t OFF_Q16  = OFF_Y16 + ELE * 2;
constexpr size_t OFF_K16  = OFF_Q16 + ELE * 2;
constexpr size_t OFF_V16  = OFF_K16 + ELE * 2;
constexpr size_t OFF_AT16 = OFF_V16 + ELE * 2;          // attention out (f16)
constexpr size_t OFF_A16  = OFF_X16;                    // gamma*attn + y (aliases X16)
constexpr size_t OFF_Z    = OFF_AT16 + ELE * 2;         // conv2+bn2 out (f32)
constexpr size_t OFF_W1H  = OFF_Z + ELE * 4;
constexpr size_t OFF_WQH  = OFF_W1H + (size_t)CC * CC * 2;
constexpr size_t OFF_WKH  = OFF_WQH + (size_t)CC * CC * 2;
constexpr size_t OFF_WVH  = OFF_WKH + (size_t)CC * CC * 2;
constexpr size_t OFF_W2H  = OFF_WVH + (size_t)CC * CC * 2;
constexpr size_t OFF_S1   = OFF_W2H + (size_t)CC * CC * 2;
constexpr size_t OFF_T1   = OFF_S1 + CC * 4;
constexpr size_t OFF_S2   = OFF_T1 + CC * 4;
constexpr size_t OFF_T2   = OFF_S2 + CC * 4;
constexpr size_t OFF_ONES = OFF_T2 + CC * 4;
constexpr size_t OFF_AVG  = OFF_ONES + CC * 4;          // [B,C]
constexpr size_t OFF_MX   = OFF_AVG + (size_t)BB * CC * 4;
constexpr size_t OFF_CA   = OFF_MX + (size_t)BB * CC * 4;
constexpr size_t OFF_AVGS = OFF_CA + (size_t)BB * CC * 4;   // [B,N]
constexpr size_t OFF_MAXS = OFF_AVGS + (size_t)BB * NS * 4;

// ---------------- prep: weights->f16, BN fold ----------------
__global__ __launch_bounds__(256)
void prep_k(const float* __restrict__ w1, const float* __restrict__ wq,
            const float* __restrict__ wk, const float* __restrict__ wv,
            const float* __restrict__ w2,
            const float* __restrict__ g1, const float* __restrict__ b1,
            const float* __restrict__ m1, const float* __restrict__ v1,
            const float* __restrict__ g2, const float* __restrict__ b2,
            const float* __restrict__ m2, const float* __restrict__ v2,
            _Float16* w1h, _Float16* wqh, _Float16* wkh, _Float16* wvh,
            _Float16* w2h, float* s1, float* t1, float* s2, float* t2,
            float* ones)
{
    int i = blockIdx.x * 256 + threadIdx.x;
    const int WSZ = CC * CC;
    if (i < WSZ)            { w1h[i] = (_Float16)w1[i]; return; }
    if (i < 2 * WSZ)        { wqh[i - WSZ] = (_Float16)wq[i - WSZ]; return; }
    if (i < 3 * WSZ)        { wkh[i - 2 * WSZ] = (_Float16)wk[i - 2 * WSZ]; return; }
    if (i < 4 * WSZ)        { wvh[i - 3 * WSZ] = (_Float16)wv[i - 3 * WSZ]; return; }
    if (i < 5 * WSZ)        { w2h[i - 4 * WSZ] = (_Float16)w2[i - 4 * WSZ]; return; }
    int j = i - 5 * WSZ;
    if (j < CC)             { float s = g1[j] * rsqrtf(v1[j] + 1e-5f); s1[j] = s; t1[j] = b1[j] - m1[j] * s; return; }
    if (j < 2 * CC)         { j -= CC; float s = g2[j] * rsqrtf(v2[j] + 1e-5f); s2[j] = s; t2[j] = b2[j] - m2[j] * s; return; }
    if (j < 3 * CC)         { ones[j - 2 * CC] = 1.0f; }
}

__global__ __launch_bounds__(256)
void cvt_x_k(const float* __restrict__ x, _Float16* __restrict__ x16)
{
    size_t i = (size_t)blockIdx.x * 256 + threadIdx.x;
    x16[i] = (_Float16)x[i];
}

// ---------------- GEMM: OUT[b,o,n] = act(scale[o]*sum_c W[o,c]*X[b,c,n] + shift[o])
// grid: (NS/64, CC/128, BB), block 256 = 8 waves.
// Each wave owns a 16(o) x 64(n) output strip: one A (weight) fragment per
// k-step feeds 4 WMMAs (4x A-reuse -> 4x less global W traffic).
__global__ __launch_bounds__(256)
void gemm_k(const _Float16* __restrict__ W, const _Float16* __restrict__ X,
            void* __restrict__ OUT,
            const float* __restrict__ scale, const float* __restrict__ shift,
            int relu, int outf32)
{
    __shared__ alignas(16) _Float16 Xl[64 * CC];   // [n][k], k fast (32 KB)
    const int b  = blockIdx.z;
    const int n0 = blockIdx.x * 64;
    const int ob = blockIdx.y * 128;
    const int t  = threadIdx.x;

    // stage X tile transposed: thread t owns input channel c=t (64 n values)
    {
        const _Float16* src = X + ((size_t)b * CC + t) * NS + n0;
#pragma unroll
        for (int j0 = 0; j0 < 64; j0 += 8) {
            union { uint4 u; _Float16 h[8]; } a;
            a.u = *(const uint4*)(src + j0);
#pragma unroll
            for (int j = 0; j < 8; j++) Xl[(j0 + j) * CC + t] = a.h[j];
        }
    }
    __syncthreads();

    const int wave = t >> 5, lane = t & 31, g = lane >> 4, ln = lane & 15;
    const int obase = ob + wave * 16;
    v8f acc[4] = {{}, {}, {}, {}};
#pragma unroll
    for (int k0 = 0; k0 < CC; k0 += 32) {
        Frag16 A;
        const _Float16* wr = W + (size_t)(obase + ln) * CC + k0 + g * 8;
        __builtin_prefetch((const void*)(wr + 32), 0, 0);
        A.q[0] = *(const uint4*)(wr);
        A.q[1] = *(const uint4*)(wr + 16);
#pragma unroll
        for (int s = 0; s < 4; s++) {
            Frag16 Bf;
            const _Float16* xr = Xl + (s * 16 + ln) * CC + k0 + g * 8;
            Bf.q[0] = *(const uint4*)(xr);
            Bf.q[1] = *(const uint4*)(xr + 16);
            acc[s] = wmma_f16(A, Bf, acc[s]);
        }
    }
#pragma unroll
    for (int s = 0; s < 4; s++) {
        const size_t outb = (size_t)b * CC * NS + n0 + s * 16 + ln;
#pragma unroll
        for (int r = 0; r < 8; r++) {
            int o = obase + r + 8 * g;
            float v = acc[s][r] * scale[o] + shift[o];
            if (relu) v = fmaxf(v, 0.0f);
            if (outf32) ((float*)OUT)[outb + (size_t)o * NS] = v;
            else        ((_Float16*)OUT)[outb + (size_t)o * NS] = (_Float16)v;
        }
    }
}

// ---------------- flash attention per (b, head, 64-row n tile) ----------------
// grid: (NS/64, BB*HEADS), block 128 = 4 waves (16 rows each)
__global__ __launch_bounds__(128)
void attn_k(const _Float16* __restrict__ Q, const _Float16* __restrict__ K,
            const _Float16* __restrict__ V, _Float16* __restrict__ O)
{
    __shared__ alignas(16) _Float16 Ql[64 * 64];        // [n][d]
    __shared__ alignas(16) _Float16 Kl[32 * 64];        // [m][d]
    __shared__ alignas(16) _Float16 Vl[64 * 32];        // [d][m]
    __shared__ alignas(16) _Float16 Pl[4][16 * 32];     // per-wave P tile [n][m]

    const int bh = blockIdx.y;
    const int b = bh >> 2, h = bh & 3;
    const int n0 = blockIdx.x * 64;
    const int t = threadIdx.x;
    const int wave = t >> 5, lane = t & 31, g = lane >> 4, ln = lane & 15;
    const size_t base = ((size_t)b * CC + h * DH) * NS;  // [b, h*64 + d, n]

    // stage Q tile transposed and pre-scaled by 1/sqrt(D)=0.125
    for (int i = t; i < 64 * 64; i += 128) {
        int n = i & 63, d = i >> 6;                      // coalesced over n
        Ql[n * 64 + d] = (_Float16)((float)Q[base + (size_t)d * NS + n0 + n] * 0.125f);
    }
    __syncthreads();

    // per-wave Q fragments (rows n = wave*16 + ln), k-chunks d=[0,32) and [32,64)
    Frag16 Aq0, Aq1;
    {
        const _Float16* qr = Ql + (wave * 16 + ln) * 64 + g * 8;
        Aq0.q[0] = *(const uint4*)(qr);
        Aq0.q[1] = *(const uint4*)(qr + 16);
        Aq1.q[0] = *(const uint4*)(qr + 32);
        Aq1.q[1] = *(const uint4*)(qr + 48);
    }

    float mrun[8], lrun[8];
    v8f oacc[4] = {{}, {}, {}, {}};
#pragma unroll
    for (int r = 0; r < 8; r++) { mrun[r] = -1e30f; lrun[r] = 0.0f; }

    for (int m0 = 0; m0 < NS; m0 += 32) {
        __syncthreads();    // previous-iteration K/V reads done before restage
        for (int i = t; i < 32 * 64; i += 128) {
            int m = i & 31, d = i >> 5;                  // coalesced over m
            Kl[m * 64 + d] = K[base + (size_t)d * NS + m0 + m];
        }
        for (int i = t; i < 64 * 32; i += 128) {
            int m = i & 31, d = i >> 5;
            Vl[d * 32 + m] = V[base + (size_t)d * NS + m0 + m];
        }
        __syncthreads();

        // scores: s0 = Q.Kt cols [m0,m0+16), s1 = cols [m0+16,m0+32)
        v8f s0 = {}, s1 = {};
        {
            Frag16 B0, B1;
            const _Float16* kr = Kl + ln * 64 + g * 8;
            B0.q[0] = *(const uint4*)(kr);      B0.q[1] = *(const uint4*)(kr + 16);
            B1.q[0] = *(const uint4*)(kr + 32); B1.q[1] = *(const uint4*)(kr + 48);
            s0 = wmma_f16(Aq0, B0, s0);
            s0 = wmma_f16(Aq1, B1, s0);
        }
        {
            Frag16 B0, B1;
            const _Float16* kr = Kl + (16 + ln) * 64 + g * 8;
            B0.q[0] = *(const uint4*)(kr);      B0.q[1] = *(const uint4*)(kr + 16);
            B1.q[0] = *(const uint4*)(kr + 32); B1.q[1] = *(const uint4*)(kr + 48);
            s1 = wmma_f16(Aq0, B0, s1);
            s1 = wmma_f16(Aq1, B1, s1);
        }

        // online softmax over 32 columns (rows r+8g, cols across 16 lanes of group)
        _Float16* pw = &Pl[wave][0];
#pragma unroll
        for (int r = 0; r < 8; r++) {
            float mx = fmaxf(s0[r], s1[r]);
            mx = fmaxf(mx, __shfl_xor(mx, 1, 32));
            mx = fmaxf(mx, __shfl_xor(mx, 2, 32));
            mx = fmaxf(mx, __shfl_xor(mx, 4, 32));
            mx = fmaxf(mx, __shfl_xor(mx, 8, 32));
            float nm   = fmaxf(mrun[r], mx);
            float corr = __expf(mrun[r] - nm);
            float p0   = __expf(s0[r] - nm);
            float p1   = __expf(s1[r] - nm);
            float rs = p0 + p1;
            rs += __shfl_xor(rs, 1, 32);
            rs += __shfl_xor(rs, 2, 32);
            rs += __shfl_xor(rs, 4, 32);
            rs += __shfl_xor(rs, 8, 32);
            lrun[r] = lrun[r] * corr + rs;
            mrun[r] = nm;
            int row = r + 8 * g;
            pw[row * 32 + ln]      = (_Float16)p0;
            pw[row * 32 + 16 + ln] = (_Float16)p1;
#pragma unroll
            for (int dt = 0; dt < 4; dt++) oacc[dt][r] *= corr;
        }

        // P as A-fragment (16x32), V as B-fragments (32x16 per d-tile)
        Frag16 Ap;
        const _Float16* pr = pw + ln * 32 + g * 8;
        Ap.q[0] = *(const uint4*)(pr);
        Ap.q[1] = *(const uint4*)(pr + 16);
#pragma unroll
        for (int dt = 0; dt < 4; dt++) {
            Frag16 Bv;
            const _Float16* vr = Vl + (dt * 16 + ln) * 32 + g * 8;
            Bv.q[0] = *(const uint4*)(vr);
            Bv.q[1] = *(const uint4*)(vr + 16);
            oacc[dt] = wmma_f16(Ap, Bv, oacc[dt]);
        }
    }

    // epilogue: rows n = n0 + wave*16 + r + 8g, cols d = dt*16 + ln
#pragma unroll
    for (int dt = 0; dt < 4; dt++) {
#pragma unroll
        for (int r = 0; r < 8; r++) {
            int d = dt * 16 + ln;
            int n = n0 + wave * 16 + r + 8 * g;
            O[base + (size_t)d * NS + n] = (_Float16)(oacc[dt][r] / lrun[r]);
        }
    }
}

// ---------------- a = gamma*attn + y ----------------
__global__ __launch_bounds__(256)
void mix_k(const _Float16* __restrict__ at, const _Float16* __restrict__ y,
           const float* __restrict__ gamma, _Float16* __restrict__ a)
{
    size_t i = (size_t)blockIdx.x * 256 + threadIdx.x;
    a[i] = (_Float16)(gamma[0] * (float)at[i] + (float)y[i]);
}

// ---------------- CBAM channel stats: one block per (b,c) row ----------------
__global__ __launch_bounds__(256)
void chanstat_k(const float* __restrict__ Z, float* __restrict__ avg, float* __restrict__ mx)
{
    int row = blockIdx.x;
    const float* p = Z + (size_t)row * NS;
    int t = threadIdx.x;
    float s = 0.0f, m = -1e30f;
    for (int j = t; j < NS; j += 256) { float v = p[j]; s += v; m = fmaxf(m, v); }
    __shared__ float ss[256], sm[256];
    ss[t] = s; sm[t] = m; __syncthreads();
    for (int k = 128; k > 0; k >>= 1) {
        if (t < k) { ss[t] += ss[t + k]; sm[t] = fmaxf(sm[t], sm[t + k]); }
        __syncthreads();
    }
    if (t == 0) { avg[row] = ss[0] * (1.0f / NS); mx[row] = sm[0]; }
}

// ---------------- CBAM channel-attention MLP: one block per batch ----------------
__global__ __launch_bounds__(256)
void ca_k(const float* __restrict__ avg, const float* __restrict__ mx,
          const float* __restrict__ w1, const float* __restrict__ w2,
          float* __restrict__ ca)
{
    __shared__ float pa[CC], pm[CC], hs[32];
    int b = blockIdx.x, c = threadIdx.x;
    pa[c] = avg[b * CC + c];
    pm[c] = mx[b * CC + c];
    __syncthreads();
    if (c < 32) {
        float sa = 0.0f, sm = 0.0f;
        for (int j = 0; j < CC; j++) {
            float w = w1[c * CC + j];
            sa += w * pa[j]; sm += w * pm[j];
        }
        hs[c] = fmaxf(sa, 0.0f) + fmaxf(sm, 0.0f);
    }
    __syncthreads();
    float o = 0.0f;
    for (int r = 0; r < 32; r++) o += w2[c * 32 + r] * hs[r];
    ca[b * CC + c] = 1.0f / (1.0f + __expf(-o));
}

// ---------------- apply channel attn + spatial stats: one thread per (b,n) ----------------
__global__ __launch_bounds__(256)
void sapply_k(float* __restrict__ Z, const float* __restrict__ ca,
              float* __restrict__ avgs, float* __restrict__ maxs)
{
    int idx = blockIdx.x * 256 + threadIdx.x;
    int b = idx >> 10, n = idx & (NS - 1);
    float* zb = Z + (size_t)b * CC * NS + n;
    const float* cb = ca + b * CC;
    float s = 0.0f, m = -1e30f;
    for (int c = 0; c < CC; c++) {
        float v = zb[(size_t)c * NS] * cb[c];
        zb[(size_t)c * NS] = v;
        s += v; m = fmaxf(m, v);
    }
    avgs[idx] = s * (1.0f / CC);
    maxs[idx] = m;
}

// ---------------- spatial 7x7 conv + final residual/relu ----------------
__global__ __launch_bounds__(256)
void final_k(const float* __restrict__ Z, const float* __restrict__ x,
             const float* __restrict__ sw, const float* __restrict__ avgs,
             const float* __restrict__ maxs, float* __restrict__ out)
{
    int idx = blockIdx.x * 256 + threadIdx.x;
    int b = idx >> 10, n = idx & (NS - 1);
    int hh = n >> 5, ww = n & 31;
    float acc = 0.0f;
#pragma unroll
    for (int i = 0; i < 7; i++) {
        int h2 = hh + i - 3;
        if ((unsigned)h2 >= 32u) continue;
#pragma unroll
        for (int j = 0; j < 7; j++) {
            int w2 = ww + j - 3;
            if ((unsigned)w2 >= 32u) continue;
            int nn = b * NS + h2 * 32 + w2;
            acc += sw[i * 7 + j] * avgs[nn] + sw[49 + i * 7 + j] * maxs[nn];
        }
    }
    float sig = 1.0f / (1.0f + __expf(-acc));
    const float* zb = Z + (size_t)b * CC * NS + n;
    const float* xb = x + (size_t)b * CC * NS + n;
    float* ob = out + (size_t)b * CC * NS + n;
    for (int c = 0; c < CC; c++) {
        float v = zb[(size_t)c * NS] * sig + xb[(size_t)c * NS];
        ob[(size_t)c * NS] = fmaxf(v, 0.0f);
    }
}

// ---------------- host launcher ----------------
extern "C" void kernel_launch(void* const* d_in, const int* in_sizes, int n_in,
                              void* d_out, int out_size, void* d_ws, size_t ws_size,
                              hipStream_t stream) {
    (void)in_sizes; (void)n_in; (void)out_size; (void)ws_size;
    const float* x     = (const float*)d_in[0];
    const float* w1    = (const float*)d_in[1];
    const float* bn1g  = (const float*)d_in[2];
    const float* bn1b  = (const float*)d_in[3];
    const float* bn1m  = (const float*)d_in[4];
    const float* bn1v  = (const float*)d_in[5];
    const float* wq    = (const float*)d_in[6];
    const float* bq    = (const float*)d_in[7];
    const float* wk    = (const float*)d_in[8];
    const float* bk    = (const float*)d_in[9];
    const float* wv    = (const float*)d_in[10];
    const float* bv    = (const float*)d_in[11];
    const float* gamma = (const float*)d_in[12];
    const float* w2    = (const float*)d_in[13];
    const float* bn2g  = (const float*)d_in[14];
    const float* bn2b  = (const float*)d_in[15];
    const float* bn2m  = (const float*)d_in[16];
    const float* bn2v  = (const float*)d_in[17];
    const float* caw1  = (const float*)d_in[18];
    const float* caw2  = (const float*)d_in[19];
    const float* saw   = (const float*)d_in[20];

    char* ws = (char*)d_ws;
    _Float16* X16  = (_Float16*)(ws + OFF_X16);
    _Float16* Y16  = (_Float16*)(ws + OFF_Y16);
    _Float16* Q16  = (_Float16*)(ws + OFF_Q16);
    _Float16* K16  = (_Float16*)(ws + OFF_K16);
    _Float16* V16  = (_Float16*)(ws + OFF_V16);
    _Float16* AT16 = (_Float16*)(ws + OFF_AT16);
    _Float16* A16  = (_Float16*)(ws + OFF_A16);
    float*    Z    = (float*)(ws + OFF_Z);
    _Float16* W1H  = (_Float16*)(ws + OFF_W1H);
    _Float16* WQH  = (_Float16*)(ws + OFF_WQH);
    _Float16* WKH  = (_Float16*)(ws + OFF_WKH);
    _Float16* WVH  = (_Float16*)(ws + OFF_WVH);
    _Float16* W2H  = (_Float16*)(ws + OFF_W2H);
    float* S1   = (float*)(ws + OFF_S1);
    float* T1   = (float*)(ws + OFF_T1);
    float* S2   = (float*)(ws + OFF_S2);
    float* T2   = (float*)(ws + OFF_T2);
    float* ONES = (float*)(ws + OFF_ONES);
    float* AVG  = (float*)(ws + OFF_AVG);
    float* MX   = (float*)(ws + OFF_MX);
    float* CA   = (float*)(ws + OFF_CA);
    float* AVGS = (float*)(ws + OFF_AVGS);
    float* MAXS = (float*)(ws + OFF_MAXS);

    const int prepThreads = 5 * CC * CC + 3 * CC;
    prep_k<<<(prepThreads + 255) / 256, 256, 0, stream>>>(
        w1, wq, wk, wv, w2, bn1g, bn1b, bn1m, bn1v, bn2g, bn2b, bn2m, bn2v,
        W1H, WQH, WKH, WVH, W2H, S1, T1, S2, T2, ONES);

    cvt_x_k<<<(int)(ELE / 256), 256, 0, stream>>>(x, X16);

    dim3 ggrid(NS / 64, CC / 128, BB);
    // conv1 + bn1 + relu
    gemm_k<<<ggrid, 256, 0, stream>>>(W1H, X16, (void*)Y16, S1, T1, 1, 0);
    // q, k, v (+bias)
    gemm_k<<<ggrid, 256, 0, stream>>>(WQH, Y16, (void*)Q16, ONES, bq, 0, 0);
    gemm_k<<<ggrid, 256, 0, stream>>>(WKH, Y16, (void*)K16, ONES, bk, 0, 0);
    gemm_k<<<ggrid, 256, 0, stream>>>(WVH, Y16, (void*)V16, ONES, bv, 0, 0);

    attn_k<<<dim3(NS / 64, BB * HEADS), 128, 0, stream>>>(Q16, K16, V16, AT16);

    mix_k<<<(int)(ELE / 256), 256, 0, stream>>>(AT16, Y16, gamma, A16);

    // conv2 + bn2 (f32 out)
    gemm_k<<<ggrid, 256, 0, stream>>>(W2H, A16, (void*)Z, S2, T2, 0, 1);

    chanstat_k<<<BB * CC, 256, 0, stream>>>(Z, AVG, MX);
    ca_k<<<BB, 256, 0, stream>>>(AVG, MX, caw1, caw2, CA);
    sapply_k<<<BB * NS / 256, 256, 0, stream>>>(Z, CA, AVGS, MAXS);
    final_k<<<BB * NS / 256, 256, 0, stream>>>(Z, x, saw, AVGS, MAXS, (float*)d_out);
}
